// RefineLayer_30270929502713
// MI455X (gfx1250) — matrix operand order repaired and checked by reading, not compiled, not run on hardware
//
#include <hip/hip_runtime.h>
#include <math.h>

typedef __attribute__((ext_vector_type(16))) __bf16 v16bf;
typedef __attribute__((ext_vector_type(8)))  float  v8f;

// Problem constants (from reference setup_inputs)
#define BB   2
#define NN   8192
#define HH   32
#define WW   64
#define MM   2048            // H*W
#define CC   256
#define NS   16
#define TOPK 8

// ---------------------------------------------------------------------------
// f32 -> bf16 weight convert with zero-padding of the K dimension to Kpad.
// dst layout: [co][Kpad]; k >= K zero-filled so the GEMM needs no B predicate.
// ---------------------------------------------------------------------------
__global__ void k_f2bf_pad(const float* __restrict__ s, __bf16* __restrict__ d,
                           int K, int Kpad, int total) {
    for (int i = blockIdx.x * blockDim.x + threadIdx.x; i < total; i += gridDim.x * blockDim.x) {
        int co = i / Kpad;
        int k  = i - co * Kpad;
        d[i] = (k < K) ? (__bf16)s[(size_t)co * K + k] : (__bf16)0.0f;
    }
}

// concat(query_img_feat, pre_feat) -> bf16  layout (b*512+c)*2048+p
__global__ void k_concat_bf(const float* __restrict__ a, const float* __restrict__ b,
                            __bf16* __restrict__ d) {
    int tid = blockIdx.x * blockDim.x + threadIdx.x;
    if (tid >= BB * 512 * MM) return;
    int p = tid & (MM - 1);
    int r = tid >> 11;              // b*512 + c
    int c = r & 511;
    int bb = r >> 9;
    float v = (c < 256) ? a[(((size_t)(bb * 256 + c)) << 11) + p]
                        : b[(((size_t)(bb * 256 + (c - 256))) << 11) + p];
    d[tid] = (__bf16)v;
}

// ---------------------------------------------------------------------------
// K1: per query (b,m): ball-query first-16 (ordered, via ballot), top-8 nearest
//     to query_X_world, and writes this query's private row of weights_all_gt
//     (zero + ones fused; rows are disjoint so no cross-kernel sync needed).
// One wave per query. 8 waves / block. grid = 4096/8 = 512 blocks.
// ---------------------------------------------------------------------------
__global__ void k_select(const float* __restrict__ xyz, const float* __restrict__ nxyz,
                         const float* __restrict__ qxw,
                         int* __restrict__ idx16, int* __restrict__ cnts,
                         float* __restrict__ wout) {
    __shared__ float sD[8][256];
    __shared__ int   sI[8][256];
    __shared__ int   sBall[8][16];

    int w = threadIdx.x >> 5, lane = threadIdx.x & 31;
    int q = blockIdx.x * 8 + w;          // 0..4095
    int b = q >> 11;
    int m = q & (MM - 1);

    float cx = nxyz[q * 3 + 0], cy = nxyz[q * 3 + 1], cz = nxyz[q * 3 + 2];
    float qx = qxw[((size_t)(b * 3 + 0) << 11) + m];
    float qy = qxw[((size_t)(b * 3 + 1) << 11) + m];
    float qz = qxw[((size_t)(b * 3 + 2) << 11) + m];

    float bd[8]; int bi[8];
#pragma unroll
    for (int j = 0; j < 8; ++j) { bd[j] = 3.0e38f; bi[j] = 0; }

    int cnt = 0;
    const float* xb = xyz + (size_t)b * NN * 3;
    for (int it = 0; it < NN / 32; ++it) {
        int n = it * 32 + lane;
        float px = xb[n * 3 + 0], py = xb[n * 3 + 1], pz = xb[n * 3 + 2];
        float dx = px - cx, dy = py - cy, dz = pz - cz;
        float d2 = dx * dx + dy * dy + dz * dz;
        bool valid = d2 < 0.25f;                       // RADIUS^2
        unsigned mask = (unsigned)__ballot(valid);
        if (valid) {
            int slot = cnt + __popc(mask & ((1u << lane) - 1u));
            if (slot < NS) sBall[w][slot] = n;
        }
        cnt += __popc(mask);

        float ex = px - qx, ey = py - qy, ez = pz - qz;
        float dq = sqrtf(ex * ex + ey * ey + ez * ez);
        if (dq < bd[7]) {
            bd[7] = dq; bi[7] = n;
#pragma unroll
            for (int p = 7; p > 0; --p)
                if (bd[p] < bd[p - 1]) {
                    float t = bd[p]; bd[p] = bd[p - 1]; bd[p - 1] = t;
                    int  ti = bi[p]; bi[p] = bi[p - 1]; bi[p - 1] = ti;
                }
        }
    }
#pragma unroll
    for (int j = 0; j < 8; ++j) { sD[w][lane * 8 + j] = bd[j]; sI[w][lane * 8 + j] = bi[j]; }
    __syncthreads();

    if (lane < NS) {
        int cc = cnt < NS ? cnt : NS;
        int v = (lane < cc) ? sBall[w][lane] : (cnt > 0 ? sBall[w][0] : 0);
        idx16[q * NS + lane] = v;
    }
    if (lane == 0) {
        cnts[q] = cnt;
        float md[8]; int mi[8];
#pragma unroll
        for (int j = 0; j < 8; ++j) { md[j] = 3.0e38f; mi[j] = 0; }
        for (int t = 0; t < 256; ++t) {
            float d = sD[w][t]; int n2 = sI[w][t];
            if (d < md[7]) {
                md[7] = d; mi[7] = n2;
#pragma unroll
                for (int p = 7; p > 0; --p)
                    if (md[p] < md[p - 1]) {
                        float tt = md[p]; md[p] = md[p - 1]; md[p - 1] = tt;
                        int  ti = mi[p]; mi[p] = mi[p - 1]; mi[p - 1] = ti;
                    }
            }
        }
#pragma unroll
        for (int j = 0; j < 8; ++j) sI[w][j] = mi[j];
    }
    __syncthreads();

    int t0 = sI[w][0], t1 = sI[w][1], t2 = sI[w][2], t3 = sI[w][3];
    int t4 = sI[w][4], t5 = sI[w][5], t6 = sI[w][6], t7 = sI[w][7];
    float* row = wout + (size_t)q * NN;
    for (int t = 0; t < NN / 32; ++t) {
        int n = t * 32 + lane;
        float v = (n == t0 || n == t1 || n == t2 || n == t3 ||
                   n == t4 || n == t5 || n == t6 || n == t7) ? 1.0f : 0.0f;
        row[n] = v;
    }
}

// ---------------------------------------------------------------------------
// K2: context attention. One wave per query. Writes confid (d_out) and the
//     259 "sumed" channels as bf16 into X1[channels 256..514].
// ---------------------------------------------------------------------------
__global__ void k_attn(const float* __restrict__ qif, const float* __restrict__ feats,
                       const float* __restrict__ xyz, const float* __restrict__ nxyz,
                       const int* __restrict__ idx16, const int* __restrict__ cnts,
                       float* __restrict__ confid, __bf16* __restrict__ X1) {
    int w = threadIdx.x >> 5, lane = threadIdx.x & 31;
    int q = blockIdx.x * 8 + w;
    int b = q >> 11;
    int m = q & (MM - 1);

    int idxk[NS];
#pragma unroll
    for (int k = 0; k < NS; ++k) idxk[k] = idx16[q * NS + k];
    int cnt = cnts[q];

    const float* qb = qif + (((size_t)b * CC) << 11) + m;
    float qv[8];
#pragma unroll
    for (int j = 0; j < 8; ++j) qv[j] = qb[((size_t)(lane + 32 * j)) << 11];

    const float* fb = feats + ((size_t)b * 512) * NN;
    float sk[NS];
#pragma unroll
    for (int k = 0; k < NS; ++k) sk[k] = 0.0f;
#pragma unroll
    for (int j = 0; j < 8; ++j) {
        const float* fc = fb + (size_t)(lane + 32 * j) * NN;
#pragma unroll
        for (int k = 0; k < NS; ++k) sk[k] += qv[j] * fc[idxk[k]];
    }
#pragma unroll
    for (int k = 0; k < NS; ++k)
        for (int o = 16; o > 0; o >>= 1) sk[k] += __shfl_xor(sk[k], o, 32);

    const float* xb = xyz + (size_t)b * NN * 3;
    float cx = nxyz[q * 3 + 0], cy = nxyz[q * 3 + 1], cz = nxyz[q * 3 + 2];
    float gx[NS], gy[NS], gz[NS], lg[NS];
#pragma unroll
    for (int k = 0; k < NS; ++k) {
        float px = xb[idxk[k] * 3 + 0], py = xb[idxk[k] * 3 + 1], pz = xb[idxk[k] * 3 + 2];
        gx[k] = px; gy[k] = py; gz[k] = pz;
        float dx = px - cx, dy = py - cy, dz = pz - cz;
        float gd = sqrtf(dx * dx + dy * dy + dz * dz + 1e-8f);
        lg[k] = (k < cnt) ? sk[k] * 0.0625f - gd : -1.0e9f;   // /sqrt(256)=1/16
    }
    float mx = -3.0e38f;
#pragma unroll
    for (int k = 0; k < NS; ++k) mx = fmaxf(mx, lg[k]);
    float conf[NS]; float ssum = 0.0f;
#pragma unroll
    for (int k = 0; k < NS; ++k) { conf[k] = __expf(lg[k] - mx); ssum += conf[k]; }
    float inv = 1.0f / ssum;
#pragma unroll
    for (int k = 0; k < NS; ++k) conf[k] *= inv;

    if (lane == 0) {
        float* co = confid + (size_t)q * NS;
#pragma unroll
        for (int k = 0; k < NS; ++k) co[k] = conf[k];
    }

    // sumed geo channels (0..2 of the 259)
    float sgx = 0, sgy = 0, sgz = 0;
#pragma unroll
    for (int k = 0; k < NS; ++k) { sgx += conf[k] * gx[k]; sgy += conf[k] * gy[k]; sgz += conf[k] * gz[k]; }
    if (lane == 0) X1[(((size_t)(b * 515 + 256)) << 11) + m] = (__bf16)sgx;
    if (lane == 1) X1[(((size_t)(b * 515 + 257)) << 11) + m] = (__bf16)sgy;
    if (lane == 2) X1[(((size_t)(b * 515 + 258)) << 11) + m] = (__bf16)sgz;

    // sumed corres channels
#pragma unroll
    for (int j = 0; j < 8; ++j) {
        int c = lane + 32 * j;
        const float* fc = fb + (size_t)c * NN;
        float s2 = 0.0f;
#pragma unroll
        for (int k = 0; k < NS; ++k) s2 += conf[k] * fc[idxk[k]];
        X1[(((size_t)(b * 515 + 259 + c)) << 11) + m] = (__bf16)s2;
    }
}

// ---------------------------------------------------------------------------
// LDS-tiled WMMA implicit-GEMM conv (3x3 pad1 or 1x1), bf16 in, f32 accum.
// D[4096 x 256] = im2col(X)[4096 x K] * W^T[K x 256].
// Block tile 128(M) x 64(N), BK=32. 8 waves (4x2); each wave owns a 32x32
// tile -> 4 WMMAs per k-step with A/B fragment reuse. A and B are staged into
// LDS in *fragment-major* order (the exact per-lane WMMA register layout:
// A: k = e + 8*((e>>3)+lh); B: k = e + 8*((e>>3)+lh)), so compute-phase fragment loads
// are plain 32-byte LDS reads (ds_load_b128 pairs). Weights are pre-padded to
// Kpad (mult of 32) with zeros -> single aligned uint4 global load, no
// predicates in B. Fused BN (+residual)(+ReLU) epilogue.
// grid = (4096/128)*(256/64) = 128 blocks of 256 threads.
// ---------------------------------------------------------------------------
__global__ void k_conv_wmma(const __bf16* __restrict__ X, const __bf16* __restrict__ Wb,
                            const float* __restrict__ g, const float* __restrict__ bta,
                            const float* __restrict__ mu, const float* __restrict__ va,
                            const float* __restrict__ resid,
                            float* __restrict__ out_f32, __bf16* __restrict__ out_bf,
                            int Cin, int K, int Kpad, int ksize, int relu, int out_cstride) {
    __shared__ alignas(32) __bf16 sA[4096];   // 8 row-groups * 32 lanes * 16 elems
    __shared__ alignas(32) __bf16 sB[2048];   // 4 col-groups * 32 lanes * 16 elems

    int t    = threadIdx.x;
    int lane = t & 31;
    int wv   = t >> 5;            // 0..7
    int wm   = wv >> 1;           // 0..3  (M sub-position)
    int wn   = wv & 1;            // 0..1  (N sub-position)
    int lh   = lane >> 4, ll = lane & 15;

    int M0 = (blockIdx.x >> 2) * 128;
    int N0 = (blockIdx.x & 3) * 64;

    // A-staging role: row ar, k-group ag (16 k each). Lane-consecutive threads
    // touch consecutive pixels -> coalescable gathers.
    int ar  = t & 127;
    int ag  = t >> 7;             // 0..1
    int am  = M0 + ar;
    int ab  = am >> 11, ayx = am & (MM - 1), ay = ayx >> 6, ax = ayx & 63;
    const __bf16* xb = X + ((size_t)ab * Cin) * MM;
    int aoff0 = ((ar >> 4) << 9) + (((ar & 15)      ) << 4) + (ag << 3);  // chunk c2=0
    int aoff1 = ((ar >> 4) << 9) + (((ar & 15) + 16 ) << 4) + (ag << 3);  // chunk c2=1

    // B-staging role: col bco, quarter bq (8 k each).
    int bco = t >> 2;             // 0..63
    int bq  = t & 3;
    const __bf16* wrow = Wb + (size_t)(N0 + bco) * Kpad;
    int boff = ((bco >> 4) << 9) + (((bco & 15) + 16 * (bq & 1)) << 4) + ((bq >> 1) << 3);

    v8f acc[2][2];
#pragma unroll
    for (int i = 0; i < 2; ++i)
#pragma unroll
        for (int j = 0; j < 2; ++j)
            acc[i][j] = (v8f){0.f, 0.f, 0.f, 0.f, 0.f, 0.f, 0.f, 0.f};

    int ksteps = Kpad >> 5;
    for (int kb = 0; kb < ksteps; ++kb) {
        int k0 = kb * 32;

        // ---- stage A (im2col gather, fragment-major pack) ----
        union { __bf16 h[8]; uint4 v; } pk[2];
#pragma unroll
        for (int c2 = 0; c2 < 2; ++c2) {
#pragma unroll
            for (int jj = 0; jj < 8; ++jj) {
                int kk = ag * 16 + c2 * 8 + jj;
                int ka = k0 + kk;
                __bf16 av = (__bf16)0.0f;
                if (ka < K) {
                    if (ksize == 3) {
                        int ci = ka / 9;
                        int rr = ka - ci * 9;
                        int ky = rr / 3;
                        int kx = rr - ky * 3;
                        int yy = ay + ky - 1, xx = ax + kx - 1;
                        if ((unsigned)yy < (unsigned)HH && (unsigned)xx < (unsigned)WW)
                            av = xb[((size_t)ci << 11) + (yy << 6) + xx];
                    } else {
                        av = xb[((size_t)ka << 11) + ayx];
                    }
                }
                pk[c2].h[jj] = av;
            }
        }
        *(uint4*)&sA[aoff0] = pk[0].v;
        *(uint4*)&sA[aoff1] = pk[1].v;

        // ---- stage B (aligned 16B global load; Kpad pre-padded) ----
        union { __bf16 h[8]; uint4 v; } pb;
        pb.v = *(const uint4*)(wrow + k0 + bq * 8);
        *(uint4*)&sB[boff] = pb.v;

        __syncthreads();

        // ---- compute: 4 WMMAs with fragment reuse ----
        v16bf a0 = *(v16bf*)&sA[(((wm * 2 + 0)) << 9) + (lane << 4)];
        v16bf a1 = *(v16bf*)&sA[(((wm * 2 + 1)) << 9) + (lane << 4)];
        v16bf b0 = *(v16bf*)&sB[(((wn * 2 + 0)) << 9) + (lane << 4)];
        v16bf b1 = *(v16bf*)&sB[(((wn * 2 + 1)) << 9) + (lane << 4)];
        acc[0][0] = __builtin_amdgcn_wmma_f32_16x16x32_bf16(false, a0, false, b0, (short)0, acc[0][0], false, false);
        acc[0][1] = __builtin_amdgcn_wmma_f32_16x16x32_bf16(false, a0, false, b1, (short)0, acc[0][1], false, false);
        acc[1][0] = __builtin_amdgcn_wmma_f32_16x16x32_bf16(false, a1, false, b0, (short)0, acc[1][0], false, false);
        acc[1][1] = __builtin_amdgcn_wmma_f32_16x16x32_bf16(false, a1, false, b1, (short)0, acc[1][1], false, false);

        __syncthreads();
    }

    // ---- BN epilogue (+residual, +ReLU) ----
#pragma unroll
    for (int j = 0; j < 2; ++j) {
        int co = N0 + wn * 32 + j * 16 + ll;
        float inv = g[co] / sqrtf(va[co] + 1e-5f);
        float bet = bta[co] - mu[co] * inv;
#pragma unroll
        for (int i = 0; i < 2; ++i) {
#pragma unroll
            for (int r = 0; r < 8; ++r) {
                int mo = M0 + wm * 32 + i * 16 + 8 * lh + r;   // D: M = r + 8*lh
                int bo = mo >> 11, yo = mo & (MM - 1);
                float v = acc[i][j][r] * inv + bet;
                if (resid)  v += resid[(((size_t)(bo * 256 + co)) << 11) + yo];
                if (relu)   v = fmaxf(v, 0.0f);
                if (out_f32) out_f32[(((size_t)(bo * 256 + co)) << 11) + yo] = v;
                if (out_bf)  out_bf[(((size_t)(bo * out_cstride + co)) << 11) + yo] = (__bf16)v;
            }
        }
    }
}

// ---------------------------------------------------------------------------
// Final 1x1 conv: refined_xyz = out_w (3x256) * feat
// ---------------------------------------------------------------------------
__global__ void k_out1x1(const float* __restrict__ feat, const float* __restrict__ ow,
                         float* __restrict__ outr) {
    int tid = blockIdx.x * blockDim.x + threadIdx.x;
    if (tid >= BB * 3 * MM) return;
    int p = tid & (MM - 1);
    int r = tid >> 11;
    int c = r % 3;
    int b = r / 3;
    const float* fbse = feat + (((size_t)b * CC) << 11) + p;
    const float* wr = ow + c * CC;
    float s = 0.0f;
    for (int ci = 0; ci < CC; ++ci) s += fbse[(size_t)ci << 11] * wr[ci];
    outr[(((size_t)(b * 3 + c)) << 11) + p] = s;
}

// ---------------------------------------------------------------------------
extern "C" void kernel_launch(void* const* d_in, const int* in_sizes, int n_in,
                              void* d_out, int out_size, void* d_ws, size_t ws_size,
                              hipStream_t stream) {
    if (n_in < 37) return;
    const float* xyz  = (const float*)d_in[0];
    const float* nxyz = (const float*)d_in[1];
    const float* fts  = (const float*)d_in[2];
    const float* qif  = (const float*)d_in[3];
    const float* pref = (const float*)d_in[4];
    const float* qxw  = (const float*)d_in[5];
    // fusion_params: w1, bn1(g,b,m,v), w2, bn2, wd, bnd
    const float* fw1 = (const float*)d_in[6];
    const float *f1g = (const float*)d_in[7],  *f1b = (const float*)d_in[8],
                *f1m = (const float*)d_in[9],  *f1v = (const float*)d_in[10];
    const float* fw2 = (const float*)d_in[11];
    const float *f2g = (const float*)d_in[12], *f2b = (const float*)d_in[13],
                *f2m = (const float*)d_in[14], *f2v = (const float*)d_in[15];
    const float* fwd = (const float*)d_in[16];
    const float *fdg = (const float*)d_in[17], *fdb = (const float*)d_in[18],
                *fdm = (const float*)d_in[19], *fdv = (const float*)d_in[20];
    // res_params
    const float* rw1 = (const float*)d_in[21];
    const float *r1g = (const float*)d_in[22], *r1b = (const float*)d_in[23],
                *r1m = (const float*)d_in[24], *r1v = (const float*)d_in[25];
    const float* rw2 = (const float*)d_in[26];
    const float *r2g = (const float*)d_in[27], *r2b = (const float*)d_in[28],
                *r2m = (const float*)d_in[29], *r2v = (const float*)d_in[30];
    const float* rwd = (const float*)d_in[31];
    const float *rdg = (const float*)d_in[32], *rdb = (const float*)d_in[33],
                *rdm = (const float*)d_in[34], *rdv = (const float*)d_in[35];
    const float* ow = (const float*)d_in[36];

    // d_out regions (floats): refined | feat | confid | weights
    float* out  = (float*)d_out;
    float* o_refined = out;
    float* o_feat    = out + 12288;
    float* o_confid  = out + 12288 + 1048576;
    float* o_weights = out + 12288 + 1048576 + 65536;

    // padded K dims (mult of 32)
    const int Kf1 = 512 * 9,  Kf1p = 4608;
    const int Kf2 = 256 * 9,  Kf2p = 2304;
    const int Kfd = 512,      Kfdp = 512;
    const int Kr1 = 515 * 9,  Kr1p = 4640;
    const int Kr2 = 256 * 9,  Kr2p = 2304;
    const int Krd = 515,      Krdp = 544;

    // workspace carve-up
    char* ws = (char*)d_ws;
    size_t off = 0;
    auto alloc = [&](size_t bytes) { size_t o = off; off += (bytes + 255) & ~(size_t)255; return o; };
    size_t o_idx16 = alloc((size_t)BB * MM * NS * 4);
    size_t o_cnt   = alloc((size_t)BB * MM * 4);
    size_t o_X0    = alloc((size_t)BB * 512 * MM * 2);
    size_t o_T1    = alloc((size_t)BB * 256 * MM * 2);
    size_t o_Rf    = alloc((size_t)BB * 256 * MM * 4);
    size_t o_X1    = alloc((size_t)BB * 515 * MM * 2);
    size_t o_U1    = alloc((size_t)BB * 256 * MM * 2);
    size_t o_R2    = alloc((size_t)BB * 256 * MM * 4);
    size_t o_fw1   = alloc((size_t)256 * Kf1p * 2);
    size_t o_fw2   = alloc((size_t)256 * Kf2p * 2);
    size_t o_fwd   = alloc((size_t)256 * Kfdp * 2);
    size_t o_rw1   = alloc((size_t)256 * Kr1p * 2);
    size_t o_rw2   = alloc((size_t)256 * Kr2p * 2);
    size_t o_rwd   = alloc((size_t)256 * Krdp * 2);
    if (off > ws_size) return;

    int*    idx16 = (int*)(ws + o_idx16);
    int*    cnts  = (int*)(ws + o_cnt);
    __bf16* X0    = (__bf16*)(ws + o_X0);
    __bf16* T1    = (__bf16*)(ws + o_T1);
    float*  Rf    = (float*)(ws + o_Rf);
    __bf16* X1    = (__bf16*)(ws + o_X1);
    __bf16* U1    = (__bf16*)(ws + o_U1);
    float*  R2    = (float*)(ws + o_R2);
    __bf16* fw1b  = (__bf16*)(ws + o_fw1);
    __bf16* fw2b  = (__bf16*)(ws + o_fw2);
    __bf16* fwdb  = (__bf16*)(ws + o_fwd);
    __bf16* rw1b  = (__bf16*)(ws + o_rw1);
    __bf16* rw2b  = (__bf16*)(ws + o_rw2);
    __bf16* rwdb  = (__bf16*)(ws + o_rwd);

    // 1) weight conversions to padded bf16
    auto cv = [&](const float* s, __bf16* d, int K, int Kp) {
        int total = 256 * Kp;
        k_f2bf_pad<<<(total + 255) / 256, 256, 0, stream>>>(s, d, K, Kp, total);
    };
    cv(fw1, fw1b, Kf1, Kf1p);
    cv(fw2, fw2b, Kf2, Kf2p);
    cv(fwd, fwdb, Kfd, Kfdp);
    cv(rw1, rw1b, Kr1, Kr1p);
    cv(rw2, rw2b, Kr2, Kr2p);
    cv(rwd, rwdb, Krd, Krdp);

    // 2) X0 = concat(query_img_feat, pre_feat) as bf16
    k_concat_bf<<<(BB * 512 * MM + 255) / 256, 256, 0, stream>>>(qif, pref, X0);

    // 3) ball query + top-8 + weights_all_gt rows
    k_select<<<512, 256, 0, stream>>>(xyz, nxyz, qxw, idx16, cnts, o_weights);

    // 4) context attention -> confid + X1[256..514]
    k_attn<<<512, 256, 0, stream>>>(qif, fts, xyz, nxyz, idx16, cnts, o_confid, X1);

    // 5) fusion basic block
    k_conv_wmma<<<128, 256, 0, stream>>>(X0, fw1b, f1g, f1b, f1m, f1v,
                                         nullptr, nullptr, T1, 512, Kf1, Kf1p, 3, 1, 256);
    k_conv_wmma<<<128, 256, 0, stream>>>(X0, fwdb, fdg, fdb, fdm, fdv,
                                         nullptr, Rf, nullptr, 512, Kfd, Kfdp, 1, 0, 256);
    k_conv_wmma<<<128, 256, 0, stream>>>(T1, fw2b, f2g, f2b, f2m, f2v,
                                         Rf, nullptr, X1, 256, Kf2, Kf2p, 3, 1, 515);

    // 6) res basic block (X1 = concat(fused, sumed_map), 515 ch)
    k_conv_wmma<<<128, 256, 0, stream>>>(X1, rw1b, r1g, r1b, r1m, r1v,
                                         nullptr, nullptr, U1, 515, Kr1, Kr1p, 3, 1, 256);
    k_conv_wmma<<<128, 256, 0, stream>>>(X1, rwdb, rdg, rdb, rdm, rdv,
                                         nullptr, R2, nullptr, 515, Krd, Krdp, 1, 0, 256);
    k_conv_wmma<<<128, 256, 0, stream>>>(U1, rw2b, r2g, r2b, r2m, r2v,
                                         R2, o_feat, nullptr, 256, Kr2, Kr2p, 3, 1, 256);

    // 7) refined_xyz = 1x1 conv(feat, out_w)
    k_out1x1<<<(BB * 3 * MM + 255) / 256, 256, 0, stream>>>(o_feat, ow, o_refined);

    (void)in_sizes; (void)out_size;
}
